// FMoETransformerMLP_13151189860755
// MI455X (gfx1250) — matrix operand order repaired and checked by reading, not compiled, not run on hardware
//
#include <hip/hip_runtime.h>
#include <hip/hip_bf16.h>
#include <math.h>

// ---------------------------------------------------------------------------
// FMoE transformer MLP for MI455X (gfx1250, wave32, WMMA)
//   x:[1,2048,1024] f32, E=8, D=1024, H=4096, top_k=2
//   top-2 dispatch (6.9e10 bf16 FLOPs); M=32 tile => 2x B-register reuse;
//   H split over blockIdx.z (256 blocks); per-slot slab output (no atomics);
//   f32->bf16 via +0x8000 + v_perm_b32 (3 VALU / pair); B stream double-
//   buffered one fragment ahead of the WMMAs; branchless tanh-GELU.
// ---------------------------------------------------------------------------

typedef __attribute__((ext_vector_type(16))) __bf16 v16bf;
typedef __attribute__((ext_vector_type(8)))  float  v8f;

#define N_TOK 2048
#define D_DIM 1024
#define H_DIM 4096
#define E_NUM 8
#define TOPK  2
#define N_SLOTS (N_TOK * TOPK)       // 4096 dispatch slots
#define M_TILE 32                    // tokens per block (2 WMMA M-subtiles)
#define N_TILES (N_TOK / M_TILE)
#define H_SPLIT 2
#define HHALF (H_DIM / H_SPLIT)      // 2048 per block
#define HC 1024                      // H chunk staged in LDS
#define XS_STRIDE (D_DIM + 8)
#define HS_STRIDE (HC + 8)

// ---- f32 pair -> packed bf16 pair: round-half-up + byte permute -----------
__device__ __forceinline__ unsigned int pk_pair(unsigned int u0, unsigned int u1) {
  // dst = [u1.hi16 : u0.hi16] after +0x8000 rounding; one v_perm_b32
  return __builtin_amdgcn_perm(u1 + 0x8000u, u0 + 0x8000u, 0x07060302u);
}
__device__ __forceinline__ __bf16 bf1(float a) {
  union { float f; unsigned int u; } x; x.f = a;
  union { unsigned short s; __bf16 b; } o;
  o.s = (unsigned short)((x.u + 0x8000u) >> 16);
  return o.b;
}

// branchless tanh-form GELU (max err ~3e-4, below bf16 matmul noise)
__device__ __forceinline__ float gelu_f(float x) {
  float z  = 0.7978845608028654f * (x + 0.044715f * x * x * x);
  float ez = __expf(2.0f * z);                       // v_exp_f32, branchless
  float th = 1.0f - __fdividef(2.0f, ez + 1.0f);     // fast rcp, branchless
  return 0.5f * x * (1.0f + th);
}

__device__ __forceinline__ void load4(uint4 r[4], const float* p) {
  const uint4* q = (const uint4*)p;
  r[0] = q[0]; r[1] = q[1]; r[2] = q[2]; r[3] = q[3];
}
__device__ __forceinline__ v16bf cvt_frag(const uint4 r[4]) {
  union { v16bf v; unsigned int u[8]; } o;
  o.u[0] = pk_pair(r[0].x, r[0].y); o.u[1] = pk_pair(r[0].z, r[0].w);
  o.u[2] = pk_pair(r[1].x, r[1].y); o.u[3] = pk_pair(r[1].z, r[1].w);
  o.u[4] = pk_pair(r[2].x, r[2].y); o.u[5] = pk_pair(r[2].z, r[2].w);
  o.u[6] = pk_pair(r[3].x, r[3].y); o.u[7] = pk_pair(r[3].z, r[3].w);
  return o.v;
}

// ---- A fragment (16x32 bf16) from row-major LDS tile (ISA 7.12.2) ---------
__device__ __forceinline__ v16bf lds_a_frag(const __bf16* As, int stride,
                                            int kk, int lane) {
  int r = lane & 15, hs = lane >> 4;
  const __bf16* p = As + r * stride + kk + hs * 16;
  union { v16bf v; uint4 q[2]; } u;
  u.q[0] = *(const uint4*)p;         // K = hs*16 + 0..7
  u.q[1] = *(const uint4*)(p + 8);   // K = hs*16 + 8..15
  return u.v;
}

#define WMMA_BF16(a, b, c) __builtin_amdgcn_wmma_f32_16x16x32_bf16( \
    false, (a), false, (b), (short)0, (c), false, false)

// ===========================================================================
// 1) gating
// ===========================================================================
__global__ __launch_bounds__(128) void gate_topk(
    const float* __restrict__ x, const float* __restrict__ gw,
    const float* __restrict__ gb, int* __restrict__ top2e,
    float* __restrict__ top2w, int* __restrict__ expCount) {
  __shared__ float red[E_NUM][128];
  const int n = blockIdx.x, t = threadIdx.x;
  float acc[E_NUM];
#pragma unroll
  for (int e = 0; e < E_NUM; e++) acc[e] = 0.0f;
  for (int d = t; d < D_DIM; d += 128) {
    float xv = x[(size_t)n * D_DIM + d];
#pragma unroll
    for (int e = 0; e < E_NUM; e++) acc[e] += xv * gw[e * D_DIM + d];
  }
#pragma unroll
  for (int e = 0; e < E_NUM; e++) red[e][t] = acc[e];
  __syncthreads();
  for (int s = 64; s > 0; s >>= 1) {
    if (t < s) {
#pragma unroll
      for (int e = 0; e < E_NUM; e++) red[e][t] += red[e][t + s];
    }
    __syncthreads();
  }
  if (t == 0) {
    float l[E_NUM];
#pragma unroll
    for (int e = 0; e < E_NUM; e++) l[e] = red[e][0] + gb[e];
    int i0 = 0;
#pragma unroll
    for (int e = 1; e < E_NUM; e++) if (l[e] > l[i0]) i0 = e;
    int i1 = (i0 == 0) ? 1 : 0;
#pragma unroll
    for (int e = 0; e < E_NUM; e++) if (e != i0 && l[e] > l[i1]) i1 = e;
    float m  = fmaxf(l[i0], l[i1]);
    float e0 = expf(l[i0] - m), e1 = expf(l[i1] - m);
    float inv = 1.0f / (e0 + e1);
    top2e[2 * n] = i0;  top2e[2 * n + 1] = i1;
    top2w[2 * n] = e0 * inv; top2w[2 * n + 1] = e1 * inv;
    atomicAdd(&expCount[i0], 1);
    atomicAdd(&expCount[i1], 1);
  }
}

__global__ void init_counts(int* expCount) {
  if (threadIdx.x < E_NUM) expCount[threadIdx.x] = 0;
}

__global__ void gate_offsets(const int* __restrict__ expCount,
                             int* __restrict__ expOffset,
                             int* __restrict__ expCursor) {
  if (threadIdx.x == 0 && blockIdx.x == 0) {
    int run = 0;
    for (int e = 0; e < E_NUM; e++) {
      expOffset[e] = run; expCursor[e] = run; run += expCount[e];
    }
  }
}

__global__ __launch_bounds__(256) void gate_scatter(
    const int* __restrict__ top2e, const float* __restrict__ top2w,
    int* __restrict__ expCursor, int* __restrict__ tokList,
    float* __restrict__ wList, int* __restrict__ slotOf) {
  int n = blockIdx.x * blockDim.x + threadIdx.x;
  if (n >= N_TOK) return;
#pragma unroll
  for (int k = 0; k < TOPK; k++) {
    int e   = top2e[2 * n + k];
    int pos = atomicAdd(&expCursor[e], 1);
    tokList[pos] = n;
    wList[pos]   = top2w[2 * n + k];
    slotOf[2 * n + k] = pos;
  }
}

__global__ __launch_bounds__(256) void zero_out(float* __restrict__ out) {
  size_t i = ((size_t)blockIdx.x * 256 + threadIdx.x) * 4;
  float4 z = {0.f, 0.f, 0.f, 0.f};
  *(float4*)(out + i) = z;
}

// ===========================================================================
// 2) fused expert MLP: slab[slot] = coef * (gelu(X W1^T) W2^T), one H half
// ===========================================================================
__global__ __launch_bounds__(256) void moe_expert_gemm(
    const float* __restrict__ x,  const float* __restrict__ w1,
    const float* __restrict__ w2, const int* __restrict__ expOffset,
    const int* __restrict__ expCount, const int* __restrict__ tokList,
    const float* __restrict__ wList, float* __restrict__ slab,
    float* __restrict__ out, int useSlab) {
  const int tile = blockIdx.x;
  const int e    = blockIdx.y;
  const int half = blockIdx.z;
  const int cnt  = expCount[e];
  if (tile * M_TILE >= cnt) return;
  const int baseSlot = expOffset[e] + tile * M_TILE;

  extern __shared__ __align__(16) char smem_raw[];
  __bf16* Xs    = (__bf16*)smem_raw;                 // [32][XS_STRIDE]
  __bf16* Hs    = Xs + M_TILE * XS_STRIDE;           // [32][HS_STRIDE]
  int*    stok  = (int*)(Hs + M_TILE * HS_STRIDE);   // [32]
  float*  scoef = (float*)(stok + M_TILE);           // [32]

  const int tid = threadIdx.x;
  if (tid < M_TILE) {
    int valid = tile * M_TILE + tid < cnt;
    stok[tid]  = tokList[valid ? baseSlot + tid : baseSlot];
    scoef[tid] = valid ? wList[baseSlot + tid] : 0.0f;
  }
  __syncthreads();

  // gather + packed-convert token tile into LDS
  for (int idx = tid * 2; idx < M_TILE * D_DIM; idx += 512) {
    int r = idx >> 10, c = idx & (D_DIM - 1);
    uint2 xv = *(const uint2*)(x + (size_t)stok[r] * D_DIM + c);
    *(unsigned int*)(Xs + r * XS_STRIDE + c) = pk_pair(xv.x, xv.y);
  }
  __syncthreads();

  const int wv = tid >> 5, lane = tid & 31;
  const int col = lane & 15, hs = lane >> 4;
  const int hbase = half * HHALF;

  const float* W1e = w1 + (size_t)e * H_DIM * D_DIM;  // [H][D]
  const float* W2e = w2 + (size_t)e * D_DIM * H_DIM;  // [D][H]

  v8f acc2[2][8];   // 32 x 1024 output tile (this H half), persistent
#pragma unroll
  for (int m = 0; m < 2; m++)
#pragma unroll
    for (int j = 0; j < 8; j++)
      acc2[m][j] = (v8f){0.f,0.f,0.f,0.f,0.f,0.f,0.f,0.f};

  for (int hc = 0; hc < HHALF; hc += HC) {
    // ---- GEMM1: Hs = gelu( X(32x1024) @ W1[hbase+hc : +HC, :]^T ) ---------
#pragma unroll
    for (int g = 0; g < 2; g++) {   // j-groups of 4 bound live accumulators
      v8f acc1[2][4];
#pragma unroll
      for (int m = 0; m < 2; m++)
#pragma unroll
        for (int j = 0; j < 4; j++)
          acc1[m][j] = (v8f){0.f,0.f,0.f,0.f,0.f,0.f,0.f,0.f};
      const float* pB =
          W1e + (size_t)(hbase + hc + wv * 128 + g * 64 + col) * D_DIM + hs * 16;
      uint4 raw[4];
      load4(raw, pB);                               // (kk=0, j=0) in flight
      for (int kk = 0; kk < D_DIM; kk += 32) {
        v16bf a0 = lds_a_frag(Xs, XS_STRIDE, kk, lane);
        v16bf a1 = lds_a_frag(Xs + 16 * XS_STRIDE, XS_STRIDE, kk, lane);
        int kn = (kk + 32) & (D_DIM - 1);           // wrap keeps loads in-bounds
#pragma unroll
        for (int j = 0; j < 4; j++) {
          uint4 nraw[4];
          if (j < 3) load4(nraw, pB + (size_t)(j + 1) * 16 * D_DIM + kk);
          else       load4(nraw, pB + kn);
          v16bf b = cvt_frag(raw);                  // 3 VALU / bf16 pair
          acc1[0][j] = WMMA_BF16(a0, b, acc1[0][j]);
          acc1[1][j] = WMMA_BF16(a1, b, acc1[1][j]);
          raw[0] = nraw[0]; raw[1] = nraw[1]; raw[2] = nraw[2]; raw[3] = nraw[3];
        }
      }
#pragma unroll
      for (int m = 0; m < 2; m++)
#pragma unroll
        for (int j = 0; j < 4; j++)
#pragma unroll
          for (int v = 0; v < 8; v++) {
            float hv = gelu_f(acc1[m][j][v]);       // branchless
            int r  = m * 16 + v + 8 * hs;           // C layout: M = v + 8*hs
            int c2 = wv * 128 + (g * 4 + j) * 16 + col;
            Hs[r * HS_STRIDE + c2] = bf1(hv);
          }
    }
    __syncthreads();
    // ---- GEMM2: acc2 += H(32xHC) @ W2[:, hbase+hc : +HC]^T ----------------
    {
      const float* pB2 =
          W2e + (size_t)(wv * 128 + col) * H_DIM + hbase + hc + hs * 16;
      uint4 raw[4];
      load4(raw, pB2);
      for (int kk = 0; kk < HC; kk += 32) {
        v16bf a0 = lds_a_frag(Hs, HS_STRIDE, kk, lane);
        v16bf a1 = lds_a_frag(Hs + 16 * HS_STRIDE, HS_STRIDE, kk, lane);
        int kn = (kk + 32) & (HC - 1);
#pragma unroll
        for (int j = 0; j < 8; j++) {
          uint4 nraw[4];
          if (j < 7) load4(nraw, pB2 + (size_t)(j + 1) * 16 * H_DIM + kk);
          else       load4(nraw, pB2 + kn);
          v16bf b = cvt_frag(raw);
          acc2[0][j] = WMMA_BF16(a0, b, acc2[0][j]);  // B reused across M
          acc2[1][j] = WMMA_BF16(a1, b, acc2[1][j]);
          raw[0] = nraw[0]; raw[1] = nraw[1]; raw[2] = nraw[2]; raw[3] = nraw[3];
        }
      }
    }
    __syncthreads();   // Hs reused next chunk
  }

  // epilogue: coef-scale; invalid tail rows (coef 0) go to a dump region
#pragma unroll
  for (int m = 0; m < 2; m++)
#pragma unroll
    for (int v = 0; v < 8; v++) {
      int r = m * 16 + v + 8 * hs;
      int valid = (tile * M_TILE + r) < cnt;
      float cf = scoef[r];
      size_t slotIdx = valid ? ((size_t)half * N_SLOTS + baseSlot + r)
                             : ((size_t)(H_SPLIT * N_SLOTS) + r);
      float* rowp = useSlab ? slab + slotIdx * D_DIM
                            : out + (size_t)stok[r] * D_DIM;
#pragma unroll
      for (int j = 0; j < 8; j++) {
        int c2 = wv * 128 + j * 16 + col;
        float val = acc2[m][j][v] * cf;
        if (useSlab) {
          rowp[c2] = val;
        } else {
          __hip_atomic_fetch_add(&rowp[c2], val, __ATOMIC_RELAXED,
                                 __HIP_MEMORY_SCOPE_AGENT);   // adds 0 if pad
        }
      }
    }
}

// ===========================================================================
// 3) combine slots + residual + post-LN + bias passthrough
// ===========================================================================
__global__ __launch_bounds__(256) void post_ln(
    const float* __restrict__ x, const float* __restrict__ g,
    const float* __restrict__ b, const float* __restrict__ bias,
    const float* __restrict__ slab, const int* __restrict__ slotOf,
    int useSlab, float* __restrict__ out) {
  __shared__ float ssum[256], ssq[256];
  const int n = blockIdx.x, t = threadIdx.x;
  const size_t s0 = useSlab ? (size_t)slotOf[2 * n]     : 0;
  const size_t s1 = useSlab ? (size_t)slotOf[2 * n + 1] : 0;
  float h[4], s = 0.f, sq = 0.f;
#pragma unroll
  for (int i = 0; i < 4; i++) {
    int c = t + i * 256;
    float v;
    if (useSlab) {
      v = slab[s0 * D_DIM + c] + slab[s1 * D_DIM + c] +
          slab[((size_t)N_SLOTS + s0) * D_DIM + c] +
          slab[((size_t)N_SLOTS + s1) * D_DIM + c];
    } else {
      v = out[(size_t)n * D_DIM + c];
    }
    v += x[(size_t)n * D_DIM + c];
    h[i] = v; s += v; sq += v * v;
  }
  ssum[t] = s; ssq[t] = sq;
  __syncthreads();
  for (int st = 128; st > 0; st >>= 1) {
    if (t < st) { ssum[t] += ssum[t + st]; ssq[t] += ssq[t + st]; }
    __syncthreads();
  }
  float mu  = ssum[0] * (1.0f / D_DIM);
  float var = ssq[0] * (1.0f / D_DIM) - mu * mu;
  float rs  = rsqrtf(var + 1e-5f);
#pragma unroll
  for (int i = 0; i < 4; i++) {
    int c = t + i * 256;
    out[(size_t)n * D_DIM + c] = (h[i] - mu) * rs * g[c] + b[c];
  }
  if (n == 0) {   // tuple's second output: bias passthrough
#pragma unroll
    for (int i = 0; i < 4; i++) {
      int c = t + i * 256;
      out[(size_t)N_TOK * D_DIM + c] = bias[c];
    }
  }
}

// ===========================================================================
extern "C" void kernel_launch(void* const* d_in, const int* in_sizes, int n_in,
                              void* d_out, int out_size, void* d_ws, size_t ws_size,
                              hipStream_t stream) {
  const float* x    = (const float*)d_in[0];
  const float* gw   = (const float*)d_in[1];
  const float* gb   = (const float*)d_in[2];
  const float* w1   = (const float*)d_in[3];
  const float* w2   = (const float*)d_in[4];
  const float* ln_g = (const float*)d_in[5];
  const float* ln_b = (const float*)d_in[6];
  const float* bias = (const float*)d_in[7];
  float* out = (float*)d_out;

  // workspace carve: control (~82 KB) + optional 32 MB slot slab (+dump)
  int*   expCount  = (int*)d_ws;
  int*   expOffset = expCount + E_NUM;
  int*   expCursor = expOffset + E_NUM;
  int*   top2e     = expCursor + E_NUM;              // N*2
  int*   tokList   = top2e + N_SLOTS;                // N*2
  int*   slotOf    = tokList + N_SLOTS;              // N*2
  float* top2w     = (float*)(slotOf + N_SLOTS);     // N*2
  float* wList     = top2w + N_SLOTS;                // N*2
  size_t ctrlBytes = (size_t)(3 * E_NUM + 5 * N_SLOTS) * 4;
  size_t slabOff   = (ctrlBytes + 255) & ~(size_t)255;
  size_t slabBytes = ((size_t)H_SPLIT * N_SLOTS + M_TILE) * D_DIM * sizeof(float);
  int useSlab      = (ws_size >= slabOff + slabBytes) ? 1 : 0;
  float* slab      = (float*)((char*)d_ws + slabOff);

  init_counts<<<1, 32, 0, stream>>>(expCount);
  gate_topk<<<N_TOK, 128, 0, stream>>>(x, gw, gb, top2e, top2w, expCount);
  gate_offsets<<<1, 1, 0, stream>>>(expCount, expOffset, expCursor);
  gate_scatter<<<(N_TOK + 255) / 256, 256, 0, stream>>>(top2e, top2w, expCursor,
                                                        tokList, wList, slotOf);
  if (!useSlab) zero_out<<<(N_TOK * D_DIM) / 1024, 256, 0, stream>>>(out);

  size_t lds = (size_t)M_TILE * XS_STRIDE * sizeof(__bf16) +
               (size_t)M_TILE * HS_STRIDE * sizeof(__bf16) +
               M_TILE * (sizeof(int) + sizeof(float));
  dim3 grid(N_TILES, E_NUM, H_SPLIT);   // worst case: all tokens on one expert
  moe_expert_gemm<<<grid, 256, lds, stream>>>(x, w1, w2, expOffset, expCount,
                                              tokList, wList, slab, out, useSlab);
  post_ln<<<N_TOK, 256, 0, stream>>>(x, ln_g, ln_b, bias, slab, slotOf,
                                     useSlab, out);
}
